// LSTM_15118284882241
// MI455X (gfx1250) — compile-verified, matrix operand order checked by reading, not run on hardware
//
#include <hip/hip_runtime.h>
#include <hip/hip_bf16.h>
#include <cstdint>
#include <cstddef>

// ---------------------------------------------------------------------------
// Types for CDNA5 WMMA (wave32): D(16x16 f32) = A(16x32 bf16) x B(32x16 bf16) + C
// ---------------------------------------------------------------------------
typedef __attribute__((ext_vector_type(16))) __bf16 bf16x16;
typedef __attribute__((ext_vector_type(8)))  __bf16 bf16x8;
typedef __attribute__((ext_vector_type(8)))  float  f32x8;
typedef __attribute__((ext_vector_type(4)))  int    i32x4;

static __device__ __forceinline__ unsigned short f2bf(float f) {
  unsigned u = __float_as_uint(f);
  u += 0x7FFFu + ((u >> 16) & 1u);           // round to nearest even
  return (unsigned short)(u >> 16);
}
static __device__ __forceinline__ float bf2f(unsigned short s) {
  return __uint_as_float(((unsigned)s) << 16);
}

// ---- Fast branch-free activations (v_exp_f32 / v_rcp_f32 / v_tanh_f32) ----
#define LOG2E_F 1.4426950408889634f

static __device__ __forceinline__ float fast_exp2(float x) {
#if __has_builtin(__builtin_amdgcn_exp2f)
  return __builtin_amdgcn_exp2f(x);
#else
  return exp2f(x);
#endif
}
static __device__ __forceinline__ float fast_rcp(float x) {
#if __has_builtin(__builtin_amdgcn_rcpf)
  return __builtin_amdgcn_rcpf(x);
#else
  return 1.0f / x;
#endif
}
static __device__ __forceinline__ float fast_sigmoid(float x) {
  return fast_rcp(1.0f + fast_exp2(-LOG2E_F * x));
}
static __device__ __forceinline__ float fast_tanh(float x) {
#if __has_builtin(__builtin_amdgcn_tanhf)
  return __builtin_amdgcn_tanhf(x);
#else
  // tanh(x) = 1 - 2/(exp(2x)+1); overflow of exp2 -> inf -> rcp -> 0 -> +1 (ok)
  return 1.0f - 2.0f * fast_rcp(1.0f + fast_exp2(2.0f * LOG2E_F * x));
#endif
}

// ---- Async global->LDS (CDNA5 GLOBAL_LOAD_ASYNC_TO_LDS_B128, ASYNCcnt) ----
#if defined(__HIP_DEVICE_COMPILE__) &&                                         \
    __has_builtin(__builtin_amdgcn_global_load_async_to_lds_b128) &&           \
    __has_builtin(__builtin_amdgcn_s_wait_asynccnt)
#define HAVE_ASYNC_LDS 1
#else
#define HAVE_ASYNC_LDS 0
#endif

static __device__ __forceinline__ void copy16_g2l(unsigned short* dst_lds,
                                                  const unsigned short* src_glb) {
#if HAVE_ASYNC_LDS
  __builtin_amdgcn_global_load_async_to_lds_b128(
      (__attribute__((address_space(1))) i32x4*)(src_glb),
      (__attribute__((address_space(3))) i32x4*)(dst_lds),
      /*offset=*/0, /*cpol=*/0);
#else
  *(uint4*)dst_lds = *(const uint4*)src_glb;
#endif
}
// Wait until at most N async ops remain outstanding (in-order completion means
// the N newest-issued may still be in flight; everything older has landed).
template <int N>
static __device__ __forceinline__ void copy_wait() {
#if HAVE_ASYNC_LDS
  __builtin_amdgcn_s_wait_asynccnt(N);
#endif
}

// Sense-reversing device-wide barrier over nwg workgroups.
static __device__ __forceinline__ void grid_barrier(unsigned* cnt, unsigned* gen,
                                                    unsigned nwg) {
  __threadfence();
  __syncthreads();
  if (threadIdx.x == 0) {
    unsigned g = __hip_atomic_load(gen, __ATOMIC_RELAXED, __HIP_MEMORY_SCOPE_AGENT);
    unsigned old = __hip_atomic_fetch_add(cnt, 1u, __ATOMIC_ACQ_REL, __HIP_MEMORY_SCOPE_AGENT);
    if (old == nwg - 1u) {
      __hip_atomic_store(cnt, 0u, __ATOMIC_RELEASE, __HIP_MEMORY_SCOPE_AGENT);
      __hip_atomic_fetch_add(gen, 1u, __ATOMIC_RELEASE, __HIP_MEMORY_SCOPE_AGENT);
    } else {
      while (__hip_atomic_load(gen, __ATOMIC_ACQUIRE, __HIP_MEMORY_SCOPE_AGENT) == g) {
        __builtin_amdgcn_s_sleep(8);
      }
    }
  }
  __syncthreads();
}

// ---------------------------------------------------------------------------
// Small helper kernels
// ---------------------------------------------------------------------------
__global__ void init_bar_kernel(unsigned* p) {
  if (threadIdx.x < 2) p[threadIdx.x] = 0u;
}

__global__ void cvt_bf16_kernel(const float* __restrict__ src,
                                unsigned short* __restrict__ dst, size_t n) {
  for (size_t i = (size_t)blockIdx.x * blockDim.x + threadIdx.x; i < n;
       i += (size_t)gridDim.x * blockDim.x)
    dst[i] = f2bf(src[i]);
}

// Pack [Wih | Whh] -> bf16 [1024, K] row-major (K = Din + 256); bias = bih+bhh.
__global__ void pack_weights_kernel(const float* __restrict__ Wih,
                                    const float* __restrict__ Whh,
                                    const float* __restrict__ bih,
                                    const float* __restrict__ bhh,
                                    unsigned short* __restrict__ Wc,
                                    float* __restrict__ biasc, int Din) {
  const int K = Din + 256;
  const int total = 1024 * K;
  for (int idx = blockIdx.x * blockDim.x + threadIdx.x; idx < total;
       idx += gridDim.x * blockDim.x) {
    int r = idx / K, c = idx - r * K;
    float v = (c < Din) ? Wih[(size_t)r * Din + c] : Whh[(size_t)r * 256 + (c - Din)];
    Wc[idx] = f2bf(v);
  }
  for (int r = blockIdx.x * blockDim.x + threadIdx.x; r < 1024;
       r += gridDim.x * blockDim.x)
    biasc[r] = bih[r] + bhh[r];
}

// Final FC: out[b*T+t] = dot(h[b,t,:], fcW) + fcb
__global__ void fc_kernel(const unsigned short* __restrict__ seq,
                          const float* __restrict__ W,
                          const float* __restrict__ b,
                          float* __restrict__ out, int n) {
  int i = blockIdx.x * blockDim.x + threadIdx.x;
  if (i >= n) return;
  const unsigned short* row = seq + (size_t)i * 256;
  float acc = 0.0f;
#pragma unroll 8
  for (int j = 0; j < 256; ++j) acc += bf2f(row[j]) * W[j];
  out[i] = acc + b[0];
}

// ---------------------------------------------------------------------------
// Persistent LSTM layer kernel (templated on input width for full unroll).
// Grid = 16 WGs x 256 threads (8 wave32). WG w owns hidden cols [16w, 16w+16).
// Per step: gates[128 x 64] = [x_t | h] (128 x K) @ Wc_slice^T (K x 64), bf16
// WMMA with fp32 accumulate; cell state lives in VGPRs across all timesteps.
// x_{t+1} is async-prefetched into a double-buffered LDS tile during step t,
// so only the 64KB h tile load is exposed after each grid barrier.
// ---------------------------------------------------------------------------
template <int DIN>
__global__ void __launch_bounds__(256)
lstm_layer_kernel(const unsigned short* __restrict__ in_seq,
                  const unsigned short* __restrict__ Wc,    // [1024, K] bf16
                  const float* __restrict__ bias,           // [1024] fp32
                  unsigned short* __restrict__ out_seq,     // [128,1024,256] bf16
                  unsigned short* __restrict__ h_cur,       // [128,256] bf16
                  unsigned* __restrict__ bar_cnt, unsigned* __restrict__ bar_gen) {
  constexpr int T    = 1024;
  constexpr int NWG  = 16;
  constexpr int K    = DIN + 256;
  constexpr int KP   = K + 8;                 // W row stride (16B pad)
  constexpr int DINP = DIN + 8;               // X row stride
  constexpr int HP   = 256 + 8;               // H row stride
  constexpr int NX   = (128 * (DIN / 8)) / 256;  // per-thread X chunk loads
  extern __shared__ unsigned short smem[];
  unsigned short* Xbuf0 = smem;                          // [128][DINP]
  unsigned short* Xbuf1 = Xbuf0 + 128 * DINP;            // [128][DINP]
  unsigned short* Hbuf  = Xbuf1 + 128 * DINP;            // [128][HP]
  unsigned short* W_lds = Hbuf + 128 * HP;               // [64][KP]

  const int tid   = threadIdx.x;
  const int wg    = blockIdx.x;               // 0..15
  const int wave  = tid >> 5;                 // 0..7 -> batch row tile
  const int lane  = tid & 31;
  const int lhalf = lane >> 4;                // 0/1
  const int l16   = lane & 15;

  // ---- Load this WG's 64 gate rows (i,f,g,o blocks of 16) into LDS ----
  {
    constexpr int cpr = K >> 3;               // 16B chunks per row
    for (int idx = tid; idx < 64 * cpr; idx += 256) {
      int j = idx / cpr, ch = idx - j * cpr;
      int grow = (j >> 4) * 256 + wg * 16 + (j & 15);
      copy16_g2l(W_lds + (size_t)j * KP + ch * 8, Wc + (size_t)grow * K + ch * 8);
    }
  }
  // ---- Prefetch x_0 into Xbuf0 ----
  {
    constexpr int cpr = DIN >> 3;
    for (int idx = tid; idx < 128 * cpr; idx += 256) {
      int r = idx / cpr, c0 = (idx - r * cpr) * 8;
      copy16_g2l(Xbuf0 + (size_t)r * DINP + c0,
                 in_seq + ((size_t)r * T + 0) * DIN + c0);
    }
  }

  // Per-lane gate biases (column = wg*16 + l16)
  const float bi = bias[0 * 256 + wg * 16 + l16];
  const float bf = bias[1 * 256 + wg * 16 + l16];
  const float bg = bias[2 * 256 + wg * 16 + l16];
  const float bo = bias[3 * 256 + wg * 16 + l16];

  // Zero this WG's h chunk (h(0) = 0)
  for (int idx = tid; idx < 128 * 16; idx += 256) {
    int r = idx >> 4, c = idx & 15;
    h_cur[r * 256 + wg * 16 + c] = 0;
  }
  copy_wait<0>();                             // W + x_0 staged
  grid_barrier(bar_cnt, bar_gen, NWG);        // h(0) published

  f32x8 cc = {};                              // persistent cell-state tile

  for (int t = 0; t < T; ++t) {
    unsigned short* Xcur = (t & 1) ? Xbuf1 : Xbuf0;
    unsigned short* Xnxt = (t & 1) ? Xbuf0 : Xbuf1;

    // ---- Stage h(t) into Hbuf (critical path), then prefetch x_{t+1} ----
    {
      for (int idx = tid; idx < 128 * 32; idx += 256) {   // 256/8 = 32 chunks/row
        int r = idx >> 5, c0 = (idx & 31) * 8;
        copy16_g2l(Hbuf + (size_t)r * HP + c0, h_cur + r * 256 + c0);
      }
      if (t + 1 < T) {
        constexpr int cpr = DIN >> 3;
        for (int idx = tid; idx < 128 * cpr; idx += 256) {
          int r = idx / cpr, c0 = (idx - r * cpr) * 8;
          copy16_g2l(Xnxt + (size_t)r * DINP + c0,
                     in_seq + ((size_t)r * T + (t + 1)) * DIN + c0);
        }
        copy_wait<NX>();   // newest NX outstanding = x_{t+1}; all H loads done
      } else {
        copy_wait<0>();
      }
    }
    __syncthreads();

    // ---- GEMM: batch rows [16*wave,16*wave+16) x 4 gate tiles (i,f,g,o) ----
    f32x8 acc0 = {}, acc1 = {}, acc2 = {}, acc3 = {};
    const unsigned short* Ax = Xcur + (size_t)(wave * 16 + l16) * DINP + lhalf * 8;
    const unsigned short* Ah = Hbuf + (size_t)(wave * 16 + l16) * HP + lhalf * 8;
    const unsigned short* B0 = W_lds + (size_t)(0 * 16 + l16) * KP + lhalf * 8;
    const unsigned short* B1 = W_lds + (size_t)(1 * 16 + l16) * KP + lhalf * 8;
    const unsigned short* B2 = W_lds + (size_t)(2 * 16 + l16) * KP + lhalf * 8;
    const unsigned short* B3 = W_lds + (size_t)(3 * 16 + l16) * KP + lhalf * 8;

#define LSTM_FRAG(p)                                                           \
  __builtin_shufflevector(*(const bf16x8*)(p), *(const bf16x8*)((p) + 16),     \
                          0, 1, 2, 3, 4, 5, 6, 7, 8, 9, 10, 11, 12, 13, 14, 15)

#pragma unroll
    for (int k0 = 0; k0 < DIN; k0 += 32) {    // x-region of K
      bf16x16 a = LSTM_FRAG(Ax + k0);
      acc0 = __builtin_amdgcn_wmma_f32_16x16x32_bf16(false, a, false, LSTM_FRAG(B0 + k0),
                                                     (short)0, acc0, false, false);
      acc1 = __builtin_amdgcn_wmma_f32_16x16x32_bf16(false, a, false, LSTM_FRAG(B1 + k0),
                                                     (short)0, acc1, false, false);
      acc2 = __builtin_amdgcn_wmma_f32_16x16x32_bf16(false, a, false, LSTM_FRAG(B2 + k0),
                                                     (short)0, acc2, false, false);
      acc3 = __builtin_amdgcn_wmma_f32_16x16x32_bf16(false, a, false, LSTM_FRAG(B3 + k0),
                                                     (short)0, acc3, false, false);
    }
#pragma unroll
    for (int k0 = 0; k0 < 256; k0 += 32) {    // h-region of K
      bf16x16 a = LSTM_FRAG(Ah + k0);
      acc0 = __builtin_amdgcn_wmma_f32_16x16x32_bf16(false, a, false, LSTM_FRAG(B0 + DIN + k0),
                                                     (short)0, acc0, false, false);
      acc1 = __builtin_amdgcn_wmma_f32_16x16x32_bf16(false, a, false, LSTM_FRAG(B1 + DIN + k0),
                                                     (short)0, acc1, false, false);
      acc2 = __builtin_amdgcn_wmma_f32_16x16x32_bf16(false, a, false, LSTM_FRAG(B2 + DIN + k0),
                                                     (short)0, acc2, false, false);
      acc3 = __builtin_amdgcn_wmma_f32_16x16x32_bf16(false, a, false, LSTM_FRAG(B3 + DIN + k0),
                                                     (short)0, acc3, false, false);
    }
#undef LSTM_FRAG

    // ---- Elementwise LSTM cell update; C/D tile: row = 16*wave + v + 8*lhalf,
    //      col = wg*16 + l16. Branch-free TRANS-op activations. ----
#pragma unroll
    for (int v = 0; v < 8; ++v) {
      const int row = wave * 16 + v + 8 * lhalf;
      float iv = fast_sigmoid(acc0[v] + bi);
      float fv = fast_sigmoid(acc1[v] + bf);
      float gv = fast_tanh(acc2[v] + bg);
      float ov = fast_sigmoid(acc3[v] + bo);
      float cn = fv * cc[v] + iv * gv;
      cc[v] = cn;
      unsigned short hb = f2bf(ov * fast_tanh(cn));
      h_cur[row * 256 + wg * 16 + l16] = hb;
      out_seq[((size_t)row * T + t) * 256 + wg * 16 + l16] = hb;
    }

    grid_barrier(bar_cnt, bar_gen, NWG);      // publish h(t+1); guards LDS reuse
  }
}

// ---------------------------------------------------------------------------
// Host-side launch
// ---------------------------------------------------------------------------
extern "C" void kernel_launch(void* const* d_in, const int* in_sizes, int n_in,
                              void* d_out, int out_size, void* d_ws, size_t ws_size,
                              hipStream_t stream) {
  (void)in_sizes; (void)n_in; (void)out_size; (void)ws_size;
  const float* x    = (const float*)d_in[0];   // [128,1024,128]
  const float* Wih0 = (const float*)d_in[1];   // [1024,128]
  const float* Whh0 = (const float*)d_in[2];   // [1024,256]
  const float* WihR = (const float*)d_in[3];   // [4,1024,256]
  const float* WhhR = (const float*)d_in[4];   // [4,1024,256]
  const float* bih  = (const float*)d_in[5];   // [5,1024]
  const float* bhh  = (const float*)d_in[6];   // [5,1024]
  const float* fcW  = (const float*)d_in[7];   // [1,256]
  const float* fcb  = (const float*)d_in[8];   // [1]
  float* out = (float*)d_out;                  // [128*1024]

  char* ws = (char*)d_ws;
  size_t off = 0;
  auto alloc = [&](size_t bytes) -> void* {
    void* p = ws + off;
    off = (off + bytes + 255) & ~(size_t)255;
    return p;
  };
  unsigned short* xb    = (unsigned short*)alloc(128ull * 1024 * 128 * 2); // 32 MB
  unsigned short* seqA  = (unsigned short*)alloc(128ull * 1024 * 256 * 2); // 64 MB
  unsigned short* seqB  = (unsigned short*)alloc(128ull * 1024 * 256 * 2); // 64 MB
  unsigned short* WcBuf = (unsigned short*)alloc(5ull * 1024 * 512 * 2);   // 5 MB
  float*          biasB = (float*)alloc(5ull * 1024 * 4);
  unsigned short* h_cur = (unsigned short*)alloc(128ull * 256 * 2);
  unsigned*       bars  = (unsigned*)alloc(256);

  // LDS: 2*X + H + W (elements -> bytes)
  const size_t smem0 = (size_t)(2 * 128 * (128 + 8) + 128 * 264 + 64 * (384 + 8)) * 2;
  const size_t smem1 = (size_t)(2 * 128 * (256 + 8) + 128 * 264 + 64 * (512 + 8)) * 2;
  (void)hipFuncSetAttribute((const void*)lstm_layer_kernel<128>,
                            hipFuncAttributeMaxDynamicSharedMemorySize, (int)smem0);
  (void)hipFuncSetAttribute((const void*)lstm_layer_kernel<256>,
                            hipFuncAttributeMaxDynamicSharedMemorySize, (int)smem1);

  init_bar_kernel<<<1, 32, 0, stream>>>(bars);
  cvt_bf16_kernel<<<4096, 256, 0, stream>>>(x, xb, 128ull * 1024 * 128);

  pack_weights_kernel<<<1024, 256, 0, stream>>>(Wih0, Whh0, bih, bhh,
                                                WcBuf, biasB, 128);
  for (int l = 0; l < 4; ++l)
    pack_weights_kernel<<<1024, 256, 0, stream>>>(
        WihR + (size_t)l * 1024 * 256, WhhR + (size_t)l * 1024 * 256,
        bih + (size_t)(l + 1) * 1024, bhh + (size_t)(l + 1) * 1024,
        WcBuf + (size_t)(l + 1) * 1024 * 512, biasB + (size_t)(l + 1) * 1024, 256);

  lstm_layer_kernel<128><<<16, 256, smem0, stream>>>(xb, WcBuf, biasB,
                                                     seqA, h_cur, bars + 0, bars + 1);
  lstm_layer_kernel<256><<<16, 256, smem1, stream>>>(seqA,
                                                     WcBuf + 1ull * 1024 * 512, biasB + 1024,
                                                     seqB, h_cur, bars + 0, bars + 1);
  lstm_layer_kernel<256><<<16, 256, smem1, stream>>>(seqB,
                                                     WcBuf + 2ull * 1024 * 512, biasB + 2048,
                                                     seqA, h_cur, bars + 0, bars + 1);
  lstm_layer_kernel<256><<<16, 256, smem1, stream>>>(seqA,
                                                     WcBuf + 3ull * 1024 * 512, biasB + 3072,
                                                     seqB, h_cur, bars + 0, bars + 1);
  lstm_layer_kernel<256><<<16, 256, smem1, stream>>>(seqB,
                                                     WcBuf + 4ull * 1024 * 512, biasB + 4096,
                                                     seqA, h_cur, bars + 0, bars + 1);

  fc_kernel<<<512, 256, 0, stream>>>(seqA, fcW, fcb, out, 128 * 1024);
}
